// MultiTaskGNN_v4_78795470012789
// MI455X (gfx1250) — compile-verified, matrix operand order checked by reading, not run on hardware
//
#include <hip/hip_runtime.h>

typedef __attribute__((ext_vector_type(16))) _Float16 v16h;
typedef __attribute__((ext_vector_type(8)))  float    v8f;

#define NN 100000
#define NE 1600000
#define NG 2048

// ---------------------------------------------------------------------------
// Pack a 128x128 row-major (K x N) fp32 weight into the per-lane WMMA B-fragment
// layout for V_WMMA_F32_16X16X32_F16 (ISA 7.12.2, 16-bit B 32x16):
//   fragment id = nt*4 + ks  (nt = N/16 column tile, ks = K/32 step)
//   lane = n + 16*((kk>>3)&1), element j = ((kk>>4)&1)*8 + (kk&7)
// Each fragment is 32 lanes x 16 halves -> one contiguous 32B load per lane.
// ---------------------------------------------------------------------------
__global__ void pack_w_f16(const float* __restrict__ W, _Float16* __restrict__ out) {
  int idx = blockIdx.x * blockDim.x + threadIdx.x;
  if (idx >= 128 * 128) return;
  int K = idx >> 7, N = idx & 127;
  int nt = N >> 4, n = N & 15;
  int ks = K >> 5, kk = K & 31;
  int lane = n + (((kk >> 3) & 1) << 4);
  int j = (((kk >> 4) & 1) << 3) + (kk & 7);
  out[(size_t)(((nt * 4 + ks) * 32) + lane) * 16 + j] = (_Float16)W[idx];
}

// ---------------------------------------------------------------------------
// C[M x 128] = act( (A (+A2)) @ Bfrag + bias ),  K = N = 128, M % 16 == 0.
// One wave per 16-row strip; A fragments built once, reused for 8 col tiles.
// ---------------------------------------------------------------------------
template <bool ADD2, bool RELU, bool BIAS>
__global__ void wmma_gemm_128(const float* __restrict__ A, const float* __restrict__ A2,
                              const _Float16* __restrict__ Bf, const float* __restrict__ bias,
                              float* __restrict__ C, int M) {
  int wave = blockIdx.x * 8 + (threadIdx.x >> 5);
  int lane = threadIdx.x & 31;
  int rowBase = wave << 4;
  if (rowBase >= M) return;  // uniform per wave -> EXEC stays all-1s for WMMA

  int r = rowBase + (lane & 15);
  const float* ar  = A + (size_t)r * 128;
  const float* ar2 = ADD2 ? (A2 + (size_t)r * 128) : nullptr;
  int kb = ((lane >> 4) & 1) << 3;  // lanes 0-15: K base 0; lanes 16-31: K base 8

  // A fragments: 16-bit A 16x32 layout — lane holds row M=(lane&15),
  // halves j=0..7 -> K=kb..kb+7, j=8..15 -> K=kb+16..kb+23 (per 32-wide K step).
  v16h af[4];
#pragma unroll
  for (int ks = 0; ks < 4; ++ks) {
    int k0 = ks * 32 + kb;
#pragma unroll
    for (int j = 0; j < 8; ++j) {
      float v0 = ar[k0 + j];
      float v1 = ar[k0 + 16 + j];
      if (ADD2) { v0 += ar2[k0 + j]; v1 += ar2[k0 + 16 + j]; }
      af[ks][j]     = (_Float16)v0;
      af[ks][j + 8] = (_Float16)v1;
    }
  }

  int n  = lane & 15;
  int mo = (lane >> 4) << 3;  // C layout: VGPR v -> row rowBase + mo + v
#pragma unroll
  for (int nt = 0; nt < 8; ++nt) {
    v8f acc = {};
#pragma unroll
    for (int ks = 0; ks < 4; ++ks) {
      v16h bf = *(const v16h*)(Bf + (size_t)(((nt * 4 + ks) * 32) + lane) * 16);
      acc = __builtin_amdgcn_wmma_f32_16x16x32_f16(false, af[ks], false, bf,
                                                   (short)0, acc, false, false);
    }
    float b = BIAS ? bias[nt * 16 + n] : 0.0f;
#pragma unroll
    for (int v = 0; v < 8; ++v) {
      float o = acc[v] + b;
      if (RELU) o = fmaxf(o, 0.0f);
      C[(size_t)(rowBase + mo + v) * 128 + nt * 16 + n] = o;
    }
  }
}

// ---- GIN aggregation: agg[dst] += x[src]; deg[dst] += 1. One wave per edge. ----
__global__ void gin_agg(const float* __restrict__ x, const int* __restrict__ ei,
                        float* __restrict__ agg, float* __restrict__ deg) {
  int e = blockIdx.x * 8 + (threadIdx.x >> 5);
  int lane = threadIdx.x & 31;
  if (e >= NE) return;
  int src = ei[e];
  int dst = ei[NE + e];
  float4 v = ((const float4*)(x + (size_t)src * 128))[lane];  // coalesced 512B/wave
  float* ad = agg + (size_t)dst * 128 + lane * 4;
  atomicAdd(ad + 0, v.x); atomicAdd(ad + 1, v.y);
  atomicAdd(ad + 2, v.z); atomicAdd(ad + 3, v.w);
  if (lane == 0) atomicAdd(deg + dst, 1.0f);
}

__global__ void make_dinv(const float* __restrict__ deg, float* __restrict__ dinv) {
  int i = blockIdx.x * blockDim.x + threadIdx.x;
  if (i >= NN) return;
  dinv[i] = rsqrtf(deg[i] + 1.0f);
}

// ---- GCN message: msg[dst] += dinv[src]*dinv[dst] * hw[src]. ----
__global__ void gcn_msg(const float* __restrict__ hw, const int* __restrict__ ei,
                        const float* __restrict__ dinv, float* __restrict__ msg) {
  int e = blockIdx.x * 8 + (threadIdx.x >> 5);
  int lane = threadIdx.x & 31;
  if (e >= NE) return;
  int src = ei[e];
  int dst = ei[NE + e];
  float nrm = dinv[src] * dinv[dst];
  float4 v = ((const float4*)(hw + (size_t)src * 128))[lane];
  float* md = msg + (size_t)dst * 128 + lane * 4;
  atomicAdd(md + 0, nrm * v.x); atomicAdd(md + 1, nrm * v.y);
  atomicAdd(md + 2, nrm * v.z); atomicAdd(md + 3, nrm * v.w);
}

// ---- h = relu(msg + dinv^2 * hw + bias) ----
__global__ void gcn_combine(const float* __restrict__ msg, const float* __restrict__ hw,
                            const float* __restrict__ dinv, const float* __restrict__ bias,
                            float* __restrict__ out) {
  int idx = blockIdx.x * blockDim.x + threadIdx.x;
  if (idx >= NN * 128) return;
  int node = idx >> 7;
  int k = idx & 127;
  float di = dinv[node];
  float v = msg[idx] + di * di * hw[idx] + bias[k];
  out[idx] = fmaxf(v, 0.0f);
}

// ---- global_add_pool: g[batch[node]] += h[node]. One wave per node. ----
__global__ void pool_sum(const float* __restrict__ h, const int* __restrict__ batch,
                         float* __restrict__ g) {
  int node = blockIdx.x * 8 + (threadIdx.x >> 5);
  int lane = threadIdx.x & 31;
  if (node >= NN) return;
  int b = batch[node];
  float4 v = ((const float4*)(h + (size_t)node * 128))[lane];
  float* gd = g + (size_t)b * 128 + lane * 4;
  atomicAdd(gd + 0, v.x); atomicAdd(gd + 1, v.y);
  atomicAdd(gd + 2, v.z); atomicAdd(gd + 3, v.w);
}

// ---- out[G,3] = g2 @ lin2_w + lin2_b (tiny; one thread per output) ----
__global__ void head_out(const float* __restrict__ g2, const float* __restrict__ w,
                         const float* __restrict__ b, float* __restrict__ out) {
  int idx = blockIdx.x * blockDim.x + threadIdx.x;
  if (idx >= NG * 3) return;
  int gr = idx / 3, o = idx - gr * 3;
  float acc = b[o];
  for (int k = 0; k < 128; ++k) acc += g2[(size_t)gr * 128 + k] * w[k * 3 + o];
  out[idx] = acc;
}

extern "C" void kernel_launch(void* const* d_in, const int* in_sizes, int n_in,
                              void* d_out, int out_size, void* d_ws, size_t ws_size,
                              hipStream_t stream) {
  const float* x      = (const float*)d_in[0];
  const int*   ei     = (const int*)d_in[1];
  const int*   batch  = (const int*)d_in[2];
  const float* gin_w1 = (const float*)d_in[3];
  const float* gin_b1 = (const float*)d_in[4];
  const float* gin_w2 = (const float*)d_in[5];
  const float* gin_b2 = (const float*)d_in[6];
  const float* gcn_w  = (const float*)d_in[7];
  const float* gcn_b  = (const float*)d_in[8];
  const float* lin1_w = (const float*)d_in[9];
  const float* lin1_b = (const float*)d_in[10];
  const float* lin2_w = (const float*)d_in[11];
  const float* lin2_b = (const float*)d_in[12];
  float* out = (float*)d_out;

  const size_t NB = (size_t)NN * 128 * sizeof(float);  // 51,200,000 B
  char* ws = (char*)d_ws;
  float* agg  = (float*)(ws);               // also reused as GCN msg accumulator
  float* h1   = (float*)(ws + NB);          // also reused for post-GCN activations
  float* h2   = (float*)(ws + 2 * NB);
  float* hw   = (float*)(ws + 3 * NB);
  float* deg  = (float*)(ws + 4 * NB);
  float* dinv = (float*)(ws + 4 * NB + 524288);
  float* g    = (float*)(ws + 4 * NB + 2 * 524288);
  float* g2   = (float*)(ws + 4 * NB + 2 * 524288 + 1048576);
  _Float16* pw1 = (_Float16*)(ws + 4 * NB + 2 * 524288 + 2 * 1048576);
  _Float16* pw2 = pw1 + 16384;
  _Float16* pw3 = pw2 + 16384;
  _Float16* pw4 = pw3 + 16384;

  // Pack weights into WMMA B-fragment layout (f16).
  pack_w_f16<<<64, 256, 0, stream>>>(gin_w1, pw1);
  pack_w_f16<<<64, 256, 0, stream>>>(gin_w2, pw2);
  pack_w_f16<<<64, 256, 0, stream>>>(gcn_w,  pw3);
  pack_w_f16<<<64, 256, 0, stream>>>(lin1_w, pw4);

  // GIN: agg = scatter-add(x[src] -> dst), deg = in-degree.
  (void)hipMemsetAsync(agg, 0, NB, stream);
  (void)hipMemsetAsync(deg, 0, (size_t)NN * sizeof(float), stream);
  gin_agg<<<(NE + 7) / 8, 256, 0, stream>>>(x, ei, agg, deg);

  const int nodeStrips = NN / 16;                     // 6250 (exact)
  const int nodeBlocks = (nodeStrips + 7) / 8;        // 782
  // h1 = relu((agg + x) @ W1 + b1)
  wmma_gemm_128<true, true, true><<<nodeBlocks, 256, 0, stream>>>(agg, x, pw1, gin_b1, h1, NN);
  // h2 = relu(h1 @ W2 + b2)   (outer F.relu fused)
  wmma_gemm_128<false, true, true><<<nodeBlocks, 256, 0, stream>>>(h1, nullptr, pw2, gin_b2, h2, NN);

  make_dinv<<<(NN + 255) / 256, 256, 0, stream>>>(deg, dinv);

  // hw = h2 @ gcn_w (bias added in combine)
  wmma_gemm_128<false, false, false><<<nodeBlocks, 256, 0, stream>>>(h2, nullptr, pw3, nullptr, hw, NN);

  // GCN message passing (reuse agg buffer as msg).
  (void)hipMemsetAsync(agg, 0, NB, stream);
  gcn_msg<<<(NE + 7) / 8, 256, 0, stream>>>(hw, ei, dinv, agg);
  gcn_combine<<<(NN * 128 + 255) / 256, 256, 0, stream>>>(agg, hw, dinv, gcn_b, h1);

  // global_add_pool
  (void)hipMemsetAsync(g, 0, (size_t)NG * 128 * sizeof(float), stream);
  pool_sum<<<(NN + 7) / 8, 256, 0, stream>>>(h1, batch, g);

  // head: g2 = relu(g @ lin1_w + lin1_b); out = g2 @ lin2_w + lin2_b
  wmma_gemm_128<false, true, true><<<(NG / 16 + 7) / 8, 256, 0, stream>>>(g, nullptr, pw4, lin1_b, g2, NG);
  head_out<<<(NG * 3 + 255) / 256, 256, 0, stream>>>(g2, lin2_w, lin2_b, out);
}
